// multiHead_enc_44006234915474
// MI455X (gfx1250) — matrix operand order repaired and checked by reading, not compiled, not run on hardware
//
#include <hip/hip_runtime.h>
#include <hip/hip_bf16.h>

// ---------------------------------------------------------------------------
// Problem constants (from the reference): B=8, T=2048, E=1024, DK*H=1024
// ---------------------------------------------------------------------------
#define BATCH 8
#define SEQ   2048
#define EMB   1024
#define DMODEL 1024

typedef __attribute__((ext_vector_type(16))) _Float16 v16h;
typedef __attribute__((ext_vector_type(8)))  float    v8f;

union Frag16 { uint4 u[2]; v16h v; };

// A-fragment (16x32 f16): lane holds row m=(lane&15); halves are 16 contiguous
// K values starting at k = base + 16*(lane>>4).
__device__ __forceinline__ v16h load_afrag(const _Float16* p0) {
    Frag16 f;
    f.u[0] = *(const uint4*)(p0);
    f.u[1] = *(const uint4*)(p0 + 8);
    return f.v;
}
// B-fragment (32x16 f16): lane holds col n=(lane&15); halves are 16 contiguous
// K values starting at k = base + 16*(lane>>4).
__device__ __forceinline__ v16h load_bfrag(const _Float16* p0) {
    Frag16 f;
    f.u[0] = *(const uint4*)(p0);
    f.u[1] = *(const uint4*)(p0 + 8);
    return f.v;
}

__device__ __forceinline__ v8f wmma_f16(v16h a, v16h b, v8f c) {
    // D = A*B + C, f16 inputs, f32 accumulate
    return __builtin_amdgcn_wmma_f32_16x16x32_f16(
        /*neg_a=*/false, a, /*neg_b=*/false, b,
        /*c_mod=*/(short)0, c, /*reuse_a=*/false, /*reuse_b=*/false);
}

// Async global->LDS copy of one 16-byte chunk (gfx1250 ASYNCcnt path).
__device__ __forceinline__ void async_copy_b128(void* lds_dst, const void* gsrc) {
    uint32_t ldsaddr = (uint32_t)(uintptr_t)lds_dst;   // low 32 bits = LDS offset
    asm volatile("global_load_async_to_lds_b128 %0, %1, off"
                 :: "v"(ldsaddr), "v"(gsrc) : "memory");
}
__device__ __forceinline__ void wait_async_zero() {
    asm volatile("s_wait_asynccnt 0x0" ::: "memory");
}

// ---------------------------------------------------------------------------
// Kernel 1: fused QKV projection GEMM.
//   C[M=BATCH*SEQ, N=DMODEL] = f16( f16(x) @ f16(W) )   (f32 accumulate)
//   z = 0 -> Q (row-major), 1 -> K (row-major), 2 -> V stored transposed
//   [B][DMODEL][SEQ] so the attention P@V B-fragments are contiguous loads.
// Block: 256 thr (8 waves), tile 128x64, each wave computes 32x32.
// LDS row stride = 40 halves (80 B = 20 dwords): 16B-aligned chunks AND all
// 16 fragment-row offsets distinct mod 64 banks -> conflict-free ds_load_b128.
// ---------------------------------------------------------------------------
__global__ __launch_bounds__(256)
void qkv_gemm(const float* __restrict__ x,
              const float* __restrict__ Wq,
              const float* __restrict__ Wk,
              const float* __restrict__ Wv,
              _Float16* __restrict__ Qh,
              _Float16* __restrict__ Kh,
              _Float16* __restrict__ VTh) {
    __shared__ __align__(16) _Float16 As[128][40];  // [m][k]
    __shared__ __align__(16) _Float16 Bs[64][40];   // [n][k] (transposed for B-frags)

    const int tid  = threadIdx.x;
    const int lane = tid & 31;
    const int wv   = tid >> 5;
    const int laneLo = lane & 15;
    const int laneHi = lane >> 4;

    const int m0 = blockIdx.x * 128;
    const int n0 = blockIdx.y * 64;
    const int z  = blockIdx.z;
    const float* W = (z == 0) ? Wq : ((z == 1) ? Wk : Wv);

    const int waveM = (wv >> 1) * 32;
    const int waveN = (wv & 1) * 32;

    // Per-thread staging coordinates (fixed across k-steps).
    const int arow[4] = { (tid + 0) >> 3, (tid + 256) >> 3, (tid + 512) >> 3, (tid + 768) >> 3 };
    const int aseg    = tid & 7;
    const int brow[2] = { (tid + 0) >> 4, (tid + 256) >> 4 };
    const int bseg    = tid & 15;

    v8f acc[2][2] = {};

    for (int k0 = 0; k0 < EMB; k0 += 32) {
        __syncthreads();
        // ---- stage tiles: issue all global loads first, then convert+store ----
        float4 fa[4];
        #pragma unroll
        for (int u = 0; u < 4; ++u)
            fa[u] = *(const float4*)(x + (size_t)(m0 + arow[u]) * EMB + k0 + aseg * 4);
        float4 fb[2];
        #pragma unroll
        for (int u = 0; u < 2; ++u)
            fb[u] = *(const float4*)(W + (size_t)(k0 + brow[u]) * DMODEL + n0 + bseg * 4);

        #pragma unroll
        for (int u = 0; u < 4; ++u) {
            As[arow[u]][aseg * 4 + 0] = (_Float16)fa[u].x;
            As[arow[u]][aseg * 4 + 1] = (_Float16)fa[u].y;
            As[arow[u]][aseg * 4 + 2] = (_Float16)fa[u].z;
            As[arow[u]][aseg * 4 + 3] = (_Float16)fa[u].w;
        }
        #pragma unroll
        for (int u = 0; u < 2; ++u) {
            Bs[bseg * 4 + 0][brow[u]] = (_Float16)fb[u].x;
            Bs[bseg * 4 + 1][brow[u]] = (_Float16)fb[u].y;
            Bs[bseg * 4 + 2][brow[u]] = (_Float16)fb[u].z;
            Bs[bseg * 4 + 3][brow[u]] = (_Float16)fb[u].w;
        }
        __syncthreads();

        v16h a[2], bf[2];
        #pragma unroll
        for (int i = 0; i < 2; ++i)
            a[i] = load_afrag(&As[waveM + i * 16 + laneLo][16 * laneHi]);
        #pragma unroll
        for (int j = 0; j < 2; ++j)
            bf[j] = load_bfrag(&Bs[waveN + j * 16 + laneLo][16 * laneHi]);
        #pragma unroll
        for (int i = 0; i < 2; ++i)
            #pragma unroll
            for (int j = 0; j < 2; ++j)
                acc[i][j] = wmma_f16(a[i], bf[j], acc[i][j]);
    }

    // Epilogue. C/D layout: element r -> (m = r + 8*(lane>>4), n = lane&15).
    // Uniform z-branch hoisted OUTSIDE the store loops (one clean loop taken).
    if (z == 0) {
        #pragma unroll
        for (int i = 0; i < 2; ++i)
            #pragma unroll
            for (int j = 0; j < 2; ++j)
                #pragma unroll
                for (int r = 0; r < 8; ++r) {
                    int gm = m0 + waveM + i * 16 + r + 8 * laneHi;
                    int gn = n0 + waveN + j * 16 + laneLo;
                    Qh[(size_t)gm * DMODEL + gn] = (_Float16)acc[i][j][r];
                }
    } else if (z == 1) {
        #pragma unroll
        for (int i = 0; i < 2; ++i)
            #pragma unroll
            for (int j = 0; j < 2; ++j)
                #pragma unroll
                for (int r = 0; r < 8; ++r) {
                    int gm = m0 + waveM + i * 16 + r + 8 * laneHi;
                    int gn = n0 + waveN + j * 16 + laneLo;
                    Kh[(size_t)gm * DMODEL + gn] = (_Float16)acc[i][j][r];
                }
    } else {
        #pragma unroll
        for (int i = 0; i < 2; ++i)
            #pragma unroll
            for (int j = 0; j < 2; ++j)
                #pragma unroll
                for (int r = 0; r < 8; ++r) {
                    int gm = m0 + waveM + i * 16 + r + 8 * laneHi;
                    int gn = n0 + waveN + j * 16 + laneLo;
                    int bb = gm >> 11;            // / SEQ
                    int tt = gm & (SEQ - 1);      // % SEQ
                    VTh[((size_t)bb * DMODEL + gn) * SEQ + tt] = (_Float16)acc[i][j][r];
                }
    }
}

// ---------------------------------------------------------------------------
// Kernel 2: data-derived padding lengths, one block per batch:
//   lengths[b] = #{ s : dot(Q[b,0,:], K[b,s,:]) != 0 }   (matches reference)
// ---------------------------------------------------------------------------
__global__ __launch_bounds__(256)
void lengths_kernel(const _Float16* __restrict__ Qh,
                    const _Float16* __restrict__ Kh,
                    int* __restrict__ lengths) {
    __shared__ int red[256];
    const int b = blockIdx.x, tid = threadIdx.x;
    const _Float16* q  = Qh + (size_t)b * SEQ * DMODEL;      // row t=0
    const _Float16* kb = Kh + (size_t)b * SEQ * DMODEL;
    int cnt = 0;
    for (int s = tid; s < SEQ; s += 256) {
        const _Float16* kr = kb + (size_t)s * DMODEL;
        float acc = 0.0f;
        for (int d = 0; d < DMODEL; ++d)
            acc += (float)q[d] * (float)kr[d];
        cnt += (acc != 0.0f) ? 1 : 0;
    }
    red[tid] = cnt;
    __syncthreads();
    for (int off = 128; off > 0; off >>= 1) {
        if (tid < off) red[tid] += red[tid + off];
        __syncthreads();
    }
    if (tid == 0) lengths[b] = red[0];
}

// ---------------------------------------------------------------------------
// Kernel 3: streaming (flash-style) attention.
//   Block = (16 query rows, one batch), 8 waves.
//   Q tile staged into LDS via the gfx1250 async global->LDS path.
//   s-loop steps of 128 keys: wave w computes score tile for keys
//   [s0+16w, s0+16w+16) over the full d=1024 (32 WMMAs), then owns the
//   128-wide output slice [128w, 128w+128) of O (32 WMMAs per s-step).
// ---------------------------------------------------------------------------
__global__ __launch_bounds__(256)
void attn_kernel(const _Float16* __restrict__ Qh,
                 const _Float16* __restrict__ Kh,
                 const _Float16* __restrict__ VTh,
                 const int* __restrict__ lengths,
                 float* __restrict__ out) {
    __shared__ __align__(16) _Float16 Qs[16][1032];   // 16 query rows, full d
    __shared__ float    Ss[16][128];                  // raw scores exchange
    __shared__ __align__(16) _Float16 Ps[16][136];    // exp() probs (f16)
    __shared__ float mrow[16], lrow[16], alphaS[16];

    const int tid  = threadIdx.x;
    const int lane = tid & 31;
    const int wv   = tid >> 5;
    const int laneLo = lane & 15;
    const int laneHi = lane >> 4;

    const int t0 = blockIdx.x * 16;
    const int b  = blockIdx.y;
    const int len = lengths[b];

    const _Float16* Kb = Kh  + (size_t)b * SEQ * DMODEL;
    const _Float16* Vb = VTh + (size_t)b * DMODEL * SEQ;

    // Stage Q tile (16 x 1024 f16 = 32KB) via async global->LDS b128 copies.
    #pragma unroll
    for (int u = 0; u < 8; ++u) {
        int i = tid + u * 256;          // 0..2047 16-byte chunks
        int row = i >> 7, seg = i & 127;
        async_copy_b128(&Qs[row][seg * 8],
                        Qh + ((size_t)(b * SEQ + t0 + row)) * DMODEL + seg * 8);
    }
    if (tid < 16) { mrow[tid] = -__builtin_inff(); lrow[tid] = 0.0f; }
    wait_async_zero();
    __syncthreads();

    v8f acc[8] = {};   // O slice: 8 N-tiles of 16 cols, f32 accumulators

    for (int s0 = 0; s0 < SEQ; s0 += 128) {
        // ---- scores: S[16 x 16] for this wave's key tile ----
        v8f sacc = {};
        const int key = s0 + wv * 16 + laneLo;  // B-frag col n = lane&15
        const _Float16* kp = Kb + (size_t)key * DMODEL + laneHi * 16;
        __builtin_prefetch(kp + 128 * DMODEL, 0, 1);  // next s-tile of K
        for (int kk = 0; kk < DMODEL; kk += 32) {
            v16h a  = load_afrag(&Qs[laneLo][kk + 16 * laneHi]);
            v16h bf = load_bfrag(kp + kk);
            sacc = wmma_f16(a, bf, sacc);
        }
        // scale 1/sqrt(64) + key-padding mask, write to LDS
        #pragma unroll
        for (int r = 0; r < 8; ++r) {
            int m = r + 8 * laneHi;
            float e = sacc[r];
            Ss[m][wv * 16 + laneLo] = (key < len) ? e * 0.125f : -__builtin_inff();
        }
        __syncthreads();

        // ---- online softmax bookkeeping (wave 0, one lane per row) ----
        if (wv == 0 && lane < 16) {
            const int row = lane;
            float mold = mrow[row];
            float tmax = -__builtin_inff();
            for (int c = 0; c < 128; ++c) tmax = fmaxf(tmax, Ss[row][c]);
            float mnew  = fmaxf(mold, tmax);
            float alpha = __expf(mold - mnew);   // first tile: exp(-inf)=0, acc is 0
            float sum = 0.0f;
            for (int c = 0; c < 128; ++c) {
                float p = __expf(Ss[row][c] - mnew);
                Ps[row][c] = (_Float16)p;
                sum += p;
            }
            lrow[row]   = lrow[row] * alpha + sum;
            mrow[row]   = mnew;
            alphaS[row] = alpha;
        }
        __syncthreads();

        // ---- rescale running O, then O += P @ V  (this wave's 128 cols) ----
        #pragma unroll
        for (int nt = 0; nt < 8; ++nt)
            #pragma unroll
            for (int r = 0; r < 8; ++r)
                acc[nt][r] *= alphaS[r + 8 * laneHi];

        #pragma unroll
        for (int kt = 0; kt < 128; kt += 32) {
            v16h a = load_afrag(&Ps[laneLo][kt + 16 * laneHi]);
            #pragma unroll
            for (int nt = 0; nt < 8; ++nt) {
                int col = wv * 128 + nt * 16 + laneLo;
                const _Float16* vp = Vb + (size_t)col * SEQ + s0 + kt + laneHi * 16;
                acc[nt] = wmma_f16(a, load_bfrag(vp), acc[nt]);
            }
        }
        __syncthreads();
    }

    // ---- epilogue: divide by row sums, store f32 ----
    #pragma unroll
    for (int nt = 0; nt < 8; ++nt) {
        #pragma unroll
        for (int r = 0; r < 8; ++r) {
            int m   = r + 8 * laneHi;
            int col = wv * 128 + nt * 16 + laneLo;
            out[((size_t)(b * SEQ + t0 + m)) * DMODEL + col] = acc[nt][r] / lrow[m];
        }
    }
}

// ---------------------------------------------------------------------------
// Host launcher
// ---------------------------------------------------------------------------
extern "C" void kernel_launch(void* const* d_in, const int* in_sizes, int n_in,
                              void* d_out, int out_size, void* d_ws, size_t ws_size,
                              hipStream_t stream) {
    const float* x  = (const float*)d_in[0];
    const float* Wq = (const float*)d_in[1];
    const float* Wk = (const float*)d_in[2];
    const float* Wv = (const float*)d_in[3];
    float* out = (float*)d_out;

    const size_t QKV_BYTES = (size_t)BATCH * SEQ * DMODEL * sizeof(_Float16); // 32 MB each
    char* ws = (char*)d_ws;
    _Float16* Qh  = (_Float16*)(ws);
    _Float16* Kh  = (_Float16*)(ws + QKV_BYTES);
    _Float16* VTh = (_Float16*)(ws + 2 * QKV_BYTES);
    int*      len = (int*)(ws + 3 * QKV_BYTES);

    // 1) Q,K,V projections (V stored transposed)
    qkv_gemm<<<dim3((BATCH * SEQ) / 128, DMODEL / 64, 3), 256, 0, stream>>>(
        x, Wq, Wk, Wv, Qh, Kh, VTh);
    // 2) data-derived mask lengths
    lengths_kernel<<<dim3(BATCH), 256, 0, stream>>>(Qh, Kh, len);
    // 3) streaming attention
    attn_kernel<<<dim3(SEQ / 16, BATCH), 256, 0, stream>>>(Qh, Kh, VTh, len, out);
}
